// VectorQuantizer_SPAE_85590108274961
// MI455X (gfx1250) — matrix operand order repaired and checked by reading, not compiled, not run on hardware
//
#include <hip/hip_runtime.h>

// ---------------- CDNA5 vector types ----------------
typedef __attribute__((ext_vector_type(16))) __bf16 v16bf;
typedef __attribute__((ext_vector_type(8)))  __bf16 v8bf;
typedef __attribute__((ext_vector_type(8)))  float  v8f;

#define NPOINTS_TOTAL 8192          // 8*32*32
#define EDIM 512
#define NE 49408
#define COLTILES (NE / 16)          // 3088
#define NCHUNK 16
#define TILES_PER_CHUNK (COLTILES / NCHUNK) // 193
#define TOTAL_ELEMS (NPOINTS_TOTAL * EDIM)  // 4194304
#define SB_STRIDE 528               // 512 + 16 bf16 pad (bank spread)

static __device__ __forceinline__ __bf16 f2bf(float f) {
  unsigned u = __float_as_uint(f);
  unsigned r = (u + 0x7FFFu + ((u >> 16) & 1u)) >> 16;
  unsigned short s = (unsigned short)r;
  return __builtin_bit_cast(__bf16, s);
}

// Map masked-set index -> flat point p = b*1024 + h*32 + w
static __device__ __forceinline__ int mask_point(int layer, int i) {
  if (layer == 0) { // h in {7,23}, w in {7,23}: 4 per image
    int b = i >> 2, hh = (i >> 1) & 1, ww = i & 1;
    return (b << 10) + ((7 + (hh << 4)) << 5) + (7 + (ww << 4));
  }
  if (layer == 1) { // h = 3+4j, w = 3+4k: 64 per image
    int b = i >> 6, j = (i >> 3) & 7, k = i & 7;
    return (b << 10) + ((3 + (j << 2)) << 5) + (3 + (k << 2));
  }
  return i;
}

// ---------------- K0: emb -> bf16 + ||e||^2 ----------------
__global__ __launch_bounds__(256) void vq_prep_emb(
    const float* __restrict__ emb, __bf16* __restrict__ embb,
    float* __restrict__ esq) {
  __shared__ float sr[256];
  const int n = blockIdx.x, tid = threadIdx.x;
  const float* r = emb + (size_t)n * EDIM;
  float s = 0.f;
  #pragma unroll
  for (int c = tid; c < EDIM; c += 256) {
    float v = r[c];
    embb[(size_t)n * EDIM + c] = f2bf(v);
    s += v * v;
  }
  sr[tid] = s; __syncthreads();
  for (int st = 128; st > 0; st >>= 1) {
    if (tid < st) sr[tid] += sr[tid + st];
    __syncthreads();
  }
  if (tid == 0) esq[n] = sr[0];
}

// ---------------- K1: transpose z -> zt[p][c]; init residual & zqf ----------------
__global__ __launch_bounds__(256) void vq_prep_z(
    const float* __restrict__ z, float* __restrict__ zt,
    float* __restrict__ zr, __bf16* __restrict__ zrb,
    float* __restrict__ zqf) {
  int t = blockIdx.x * 256 + threadIdx.x;        // over 8192*512
  int c = t & (EDIM - 1);
  int p = t >> 9;
  int b = p >> 10, h = (p >> 5) & 31, w = p & 31;
  float v = z[(((size_t)b * EDIM + c) << 10) | (h << 5) | w];
  zt[t] = v; zr[t] = v; zrb[t] = f2bf(v); zqf[t] = 0.f;
}

// ---------------- K2: fused bf16 WMMA GEMM + per-row argmin ----------------
// block = 256 threads (8 waves); wave owns 16 rows; gridDim.y = column chunks.
// Software pipeline: regs hold tile t+1 while LDS buf[cur] holds tile t;
// one barrier per iteration (read buf[cur] / write buf[cur^1] never alias).
__global__ __launch_bounds__(256) void vq_argmin_gemm(
    const __bf16* __restrict__ zrb, const __bf16* __restrict__ embb,
    const float* __restrict__ esq, float2* __restrict__ cand,
    int npts, int layer) {
  __shared__ __align__(32) __bf16 sB[2][16 * SB_STRIDE];
  const int tid  = threadIdx.x;
  const int lane = tid & 31, wave = tid >> 5;
  const int hi   = lane >> 4, lo16 = lane & 15;
  const int ptBase = blockIdx.x * 128 + wave * 16;

  // ---- Preload A (this wave's 16 rows x 512 K) into 16 bf16 fragments ----
  int ptA = ptBase + lo16; if (ptA >= npts) ptA = npts - 1;
  const __bf16* arow = zrb + (size_t)mask_point(layer, ptA) * EDIM;
  v16bf afr[16];
  #pragma unroll
  for (int f = 0; f < 16; ++f) {
    const int k0 = f * 32;
    v8bf a0 = *(const v8bf*)(arow + k0 + hi * 8);        // K = k0..k0+7   (+8 hi half)
    v8bf a1 = *(const v8bf*)(arow + k0 + 16 + hi * 8);   // K = k0+16..+23 (+8 hi half)
    afr[f] = __builtin_shufflevector(a0, a1, 0,1,2,3,4,5,6,7,8,9,10,11,12,13,14,15);
  }

  float minv[8]; int mini[8];
  #pragma unroll
  for (int i = 0; i < 8; ++i) { minv[i] = 3.0e38f; mini[i] = 0x7fffffff; }

  const int t0 = blockIdx.y * TILES_PER_CHUNK;
  const int t1 = t0 + TILES_PER_CHUNK;            // uniform trip count (193 >= 2)

  // this thread's 4 copy slices of a 16KB tile (8 bf16 each)
  int crow[4], ckk[4];
  #pragma unroll
  for (int c = 0; c < 4; ++c) {
    int chunk = tid + c * 256;                    // 1024 chunks of 8 bf16
    crow[c] = chunk >> 6;                         // emb row within tile
    ckk[c]  = (chunk & 63) * 8;                   // K offset
  }

  v8bf rg[4];
  // prologue: tile t0 -> sB[0]; tile t0+1 -> regs
  #pragma unroll
  for (int c = 0; c < 4; ++c)
    rg[c] = *(const v8bf*)(embb + (size_t)t0 * 16 * EDIM + crow[c] * EDIM + ckk[c]);
  #pragma unroll
  for (int c = 0; c < 4; ++c)
    *(v8bf*)(&sB[0][crow[c] * SB_STRIDE + ckk[c]]) = rg[c];
  #pragma unroll
  for (int c = 0; c < 4; ++c)
    rg[c] = *(const v8bf*)(embb + (size_t)(t0 + 1) * 16 * EDIM + crow[c] * EDIM + ckk[c]);
  __syncthreads();

  for (int t = t0; t < t1; ++t) {
    const int cur = (t - t0) & 1;

    // ---- compute from sB[cur]: two independent WMMA chains over K halves ----
    v8f accE = {}, accO = {};
    #pragma unroll
    for (int f = 0; f < 8; ++f) {
      const __bf16* browE = &sB[cur][lo16 * SB_STRIDE + f * 32 + hi * 8];
      const __bf16* browO = &sB[cur][lo16 * SB_STRIDE + (f + 8) * 32 + hi * 8];
      v8bf b0 = *(const v8bf*)(browE);
      v8bf b1 = *(const v8bf*)(browE + 16);
      v16bf bE = __builtin_shufflevector(b0, b1, 0,1,2,3,4,5,6,7,8,9,10,11,12,13,14,15);
      v8bf b2 = *(const v8bf*)(browO);
      v8bf b3 = *(const v8bf*)(browO + 16);
      v16bf bO = __builtin_shufflevector(b2, b3, 0,1,2,3,4,5,6,7,8,9,10,11,12,13,14,15);
      accE = __builtin_amdgcn_wmma_f32_16x16x32_bf16(
          false, afr[f],     false, bE, (short)0, accE, false, false);
      accO = __builtin_amdgcn_wmma_f32_16x16x32_bf16(
          false, afr[f + 8], false, bO, (short)0, accO, false, false);
    }

    // ---- stage tile t+1 (in regs) into the other buffer; start loads of t+2 ----
    if (t + 1 < t1) {
      #pragma unroll
      for (int c = 0; c < 4; ++c)
        *(v8bf*)(&sB[cur ^ 1][crow[c] * SB_STRIDE + ckk[c]]) = rg[c];
      if (t + 2 < t1) {
        #pragma unroll
        for (int c = 0; c < 4; ++c)
          rg[c] = *(const v8bf*)(embb + (size_t)(t + 2) * 16 * EDIM + crow[c] * EDIM + ckk[c]);
      }
    }

    // ---- argmin epilogue: d = ||e||^2 - 2 z.e ----
    const int col = t * 16 + lo16;
    const float es = esq[col];
    #pragma unroll
    for (int i = 0; i < 8; ++i) {
      float d = es - 2.0f * (accE[i] + accO[i]);
      if (d < minv[i]) { minv[i] = d; mini[i] = col; }
    }

    __syncthreads();   // buf[cur^1] fully written before next iteration reads it
  }

  // ---- cross-lane argmin within 16-lane halves (C layout: VGPR i -> rows i / i+8) ----
  #pragma unroll
  for (int i = 0; i < 8; ++i) {
    float v = minv[i]; int id = mini[i];
    #pragma unroll
    for (int m = 8; m >= 1; m >>= 1) {
      float ov = __shfl_xor(v, m, 16);
      int   oi = __shfl_xor(id, m, 16);
      if (ov < v || (ov == v && oi < id)) { v = ov; id = oi; }
    }
    if (lo16 == 0) {
      int pt = ptBase + i + hi * 8;
      if (pt < npts)
        cand[(size_t)pt * NCHUNK + blockIdx.y] = make_float2(v, __int_as_float(id));
    }
  }
}

// ---------------- K3: candidate reduce + gather + masked updates + loss partials ----------------
__global__ __launch_bounds__(256) void vq_update(
    const float* __restrict__ emb, const float2* __restrict__ cand,
    const float* __restrict__ zt, float* __restrict__ zr, __bf16* __restrict__ zrb,
    float* __restrict__ zqf, float* __restrict__ lossPart,
    int* __restrict__ idxOut, int layer, int lossOff) {
  __shared__ int s_idx;
  __shared__ float sr[256];
  const int pt = blockIdx.x, tid = threadIdx.x;
  if (tid == 0) {
    float v = 3.0e38f; int id = 0x7fffffff;   // chunks ascend in column order -> first-min tie-break
    for (int c = 0; c < NCHUNK; ++c) {
      float2 cv = cand[(size_t)pt * NCHUNK + c];
      int ci = __float_as_int(cv.y);
      if (cv.x < v || (cv.x == v && ci < id)) { v = cv.x; id = ci; }
    }
    s_idx = id;
    if (layer == 2) idxOut[pt] = id;          // final-layer indices are the returned idx
  }
  __syncthreads();
  const int id = s_idx;
  const int p = mask_point(layer, pt);
  const int h = (p >> 5) & 31, w = p & 31;
  const bool in_m2 = ((h & 3) == 3) && ((w & 3) == 3);
  const bool in_m1 = ((h & 15) == 7) && ((w & 15) == 7);
  const float invd = 1.0f / (float)(1 + (in_m2 ? 1 : 0) + (in_m1 ? 1 : 0));
  const float* er = emb + (size_t)id * EDIM;
  float ls = 0.f;
  #pragma unroll
  for (int c = tid; c < EDIM; c += 256) {
    size_t o = (size_t)p * EDIM + c;
    float zq = er[c];
    float q  = zqf[o] + zq * invd;            // z_q_final += z_q * m / denom
    zqf[o] = q;
    float df = q - zt[o];                     // (zqf - zt*m)^2 at masked position
    ls += df * df;
    if (layer < 2) {                          // z_l += (zt - z_q) * m
      float nz = zr[o] + (zt[o] - zq);
      zr[o] = nz;
      zrb[o] = f2bf(nz);
    }
  }
  sr[tid] = ls; __syncthreads();
  for (int st = 128; st > 0; st >>= 1) {
    if (tid < st) sr[tid] += sr[tid + st];
    __syncthreads();
  }
  if (tid == 0) lossPart[lossOff + pt] = sr[0];
}

// ---------------- K4: deterministic loss finalize ----------------
__global__ __launch_bounds__(256) void vq_final_loss(
    const float* __restrict__ lossPart, float* __restrict__ outLoss) {
  __shared__ float sr[256];
  const int tid = threadIdx.x;
  float a = 0.f;
  for (int i = tid; i < 32;   i += 256) a += lossPart[i];
  for (int i = tid; i < 512;  i += 256) a += lossPart[32 + i];
  for (int i = tid; i < 8192; i += 256) a += lossPart[544 + i];
  sr[tid] = a; __syncthreads();
  for (int st = 128; st > 0; st >>= 1) {
    if (tid < st) sr[tid] += sr[tid + st];
    __syncthreads();
  }
  if (tid == 0) outLoss[0] = sr[0] * (1.0f / (float)TOTAL_ELEMS);
}

// ---------------- K5: out[b,c,h,w] = zqf[p,c]  (out == z_q_final numerically) ----------------
__global__ __launch_bounds__(256) void vq_out(
    const float* __restrict__ zqf, float* __restrict__ out) {
  int t = blockIdx.x * 256 + threadIdx.x;     // linear over out [b][c][h][w]
  int w = t & 31, h = (t >> 5) & 31, c = (t >> 10) & 511, b = t >> 19;
  int p = (b << 10) | (h << 5) | w;
  out[t] = zqf[(size_t)p * EDIM + c];
}

// ---------------- launch ----------------
extern "C" void kernel_launch(void* const* d_in, const int* in_sizes, int n_in,
                              void* d_out, int out_size, void* d_ws, size_t ws_size,
                              hipStream_t stream) {
  (void)in_sizes; (void)n_in; (void)out_size; (void)ws_size;
  const float* z   = (const float*)d_in[0];   // [8,512,32,32]
  const float* emb = (const float*)d_in[1];   // [49408,512]
  float* out = (float*)d_out;                 // [4194304 out | 1 loss | 8192 idx(int bits)]

  char* ws = (char*)d_ws;
  size_t off = 0;
  auto carve = [&](size_t bytes) -> char* {
    char* p = ws + off;
    off += (bytes + 255) & ~(size_t)255;
    return p;
  };
  float*  zt       = (float*) carve((size_t)TOTAL_ELEMS * 4);
  float*  zr       = (float*) carve((size_t)TOTAL_ELEMS * 4);
  float*  zqf      = (float*) carve((size_t)TOTAL_ELEMS * 4);
  __bf16* zrb      = (__bf16*)carve((size_t)TOTAL_ELEMS * 2);
  __bf16* embb     = (__bf16*)carve((size_t)NE * EDIM * 2);
  float*  esq      = (float*) carve((size_t)NE * 4);
  float2* cand     = (float2*)carve((size_t)NPOINTS_TOTAL * NCHUNK * sizeof(float2));
  float*  lossPart = (float*) carve((size_t)(32 + 512 + 8192) * 4);

  float* outLoss = out + TOTAL_ELEMS;
  int*   idxOut  = (int*)(out + TOTAL_ELEMS + 1);

  vq_prep_emb<<<NE, 256, 0, stream>>>(emb, embb, esq);
  vq_prep_z<<<TOTAL_ELEMS / 256, 256, 0, stream>>>(z, zt, zr, zrb, zqf);

  const int nptsArr[3] = {32, 512, 8192};
  const int lossOffArr[3] = {0, 32, 544};
  for (int l = 0; l < 3; ++l) {
    const int npts = nptsArr[l];
    const int rowTiles = (npts + 127) / 128;
    vq_argmin_gemm<<<dim3(rowTiles, NCHUNK), 256, 0, stream>>>(
        zrb, embb, esq, cand, npts, l);
    vq_update<<<npts, 256, 0, stream>>>(
        emb, cand, zt, zr, zrb, zqf, lossPart, idxOut, l, lossOffArr[l]);
  }

  vq_final_loss<<<1, 256, 0, stream>>>(lossPart, outLoss);
  vq_out<<<TOTAL_ELEMS / 256, 256, 0, stream>>>(zqf, out);
}
